// SDNet1_38646115730117
// MI455X (gfx1250) — matrix operand order repaired and checked by reading, not compiled, not run on hardware
//
#include <hip/hip_runtime.h>
#include <hip/hip_bf16.h>
#include <math.h>

// ---------------------------------------------------------------------------
// MI455X / gfx1250 implementation of the point-transformer refinement block.
// bf16 WMMA (v_wmma_f32_16x16x32_bf16) for all three GEMM stages:
//   (1) KNN feature-distance GEMM (fused with exact streaming top-16),
//       software-pipelined A-tile loads so WMMAs are not load-latency bound
//   (2) pos-embedding GEMM (256x64)
//   (3) attention MLP GEMMs (256x256 twice), softmax over k via lane shuffles
// ---------------------------------------------------------------------------

typedef __attribute__((ext_vector_type(16))) __bf16 v16bf;
typedef __attribute__((ext_vector_type(8)))  __bf16 v8bf;
typedef __attribute__((ext_vector_type(8)))  float  v8f;

#define EPSBN 1e-5f
#define BB 4
#define CC 256
#define NN 2048
#define MM 4096
#define KNN 16
#define PH 64
#define FLT_BIG 3.4e38f

static __device__ __forceinline__ unsigned short f2bf(float f) {
  unsigned u = __builtin_bit_cast(unsigned, f);
  u += 0x7FFFu + ((u >> 16) & 1u);          // round-to-nearest-even
  return (unsigned short)(u >> 16);
}
static __device__ __forceinline__ float bf2f(unsigned short h) {
  unsigned u = ((unsigned)h) << 16;
  return __builtin_bit_cast(float, u);
}
static __device__ __forceinline__ v16bf cmb(v8bf a, v8bf b) {
  return __builtin_shufflevector(a, b, 0,1,2,3,4,5,6,7,8,9,10,11,12,13,14,15);
}
static __device__ __forceinline__ v16bf ld16(const unsigned short* p0,
                                             const unsigned short* p1) {
  return cmb(*(const v8bf*)p0, *(const v8bf*)p1);
}
static __device__ __forceinline__ v8f wmma_bf(v16bf a, v16bf b, v8f c) {
  return __builtin_amdgcn_wmma_f32_16x16x32_bf16(false, a, false, b,
                                                 (short)0, c, false, false);
}

// --------------------------------------------------------------------------
// Kernel 0a: fusion_feat (B,C,M) f32 -> (B,M,C) bf16, plus r2[b][m]=||f||^2
// --------------------------------------------------------------------------
__global__ void k_prep_feat(const float* __restrict__ feat,
                            const float* __restrict__ featdb,
                            unsigned short* __restrict__ fus,
                            float* __restrict__ r2) {
  const int b = blockIdx.y, m = blockIdx.x, t = threadIdx.x;
  const float* src = (m < NN) ? feat : featdb;
  float v = src[((size_t)b * CC + t) * NN + (m & (NN - 1))];
  fus[((size_t)b * MM + m) * CC + t] = f2bf(v);
  float s = v * v;
  #pragma unroll
  for (int mk = 16; mk >= 1; mk >>= 1) s += __shfl_xor(s, mk, 32);
  __shared__ float red[8];
  if ((t & 31) == 0) red[t >> 5] = s;
  __syncthreads();
  if (t == 0) {
    float tot = 0.f;
    #pragma unroll
    for (int i = 0; i < 8; ++i) tot += red[i];
    r2[b * MM + m] = tot;
  }
}

// --------------------------------------------------------------------------
// Kernel 0b: f32 -> bf16 weight conversion
// --------------------------------------------------------------------------
__global__ void k_cvt_bf16(const float* __restrict__ s,
                           unsigned short* __restrict__ d, int n) {
  int i = blockIdx.x * blockDim.x + threadIdx.x;
  if (i < n) d[i] = f2bf(s[i]);
}

// --------------------------------------------------------------------------
// Kernel 1: KNN. Block = 4 waves, handles 16 queries; each wave streams 1024
// refs as 16x16 WMMA tiles (A=refs, B=queries), exact per-lane sorted top-16,
// merge 8 sublists/query in LDS, serial select-16 of 128 at the end.
// D layout: lane holds query q=lane%16; row m = mbase + r + (lane>=16 ? 8:0).
// A-tile loads are software-pipelined one K-chunk ahead of the WMMA.
// __launch_bounds__(128,1): allow ~160 VGPRs so the 8 resident query B-tiles
// plus the pipeline registers do NOT spill to scratch in the hot loop.
// --------------------------------------------------------------------------
__global__ void __launch_bounds__(128, 1)
k_knn(const unsigned short* __restrict__ fus,
      const float* __restrict__ r2g,
      int* __restrict__ idx_out) {
  __shared__ unsigned short sh_q[16][CC];   // 8 KB query tile
  __shared__ float sh_md[16 * 128];         // merge dists
  __shared__ int   sh_mi[16 * 128];         // merge indices

  const int b = blockIdx.y, nbase = blockIdx.x * 16, t = threadIdx.x;
  for (int i = t; i < 16 * CC; i += 128) {
    int q = i >> 8, c = i & (CC - 1);
    sh_q[q][c] = fus[((size_t)b * MM + nbase + q) * CC + c];
  }
  __syncthreads();

  const int wv = t >> 5, lane = t & 31, lq = lane & 15, hi = lane >> 4;
  float bd[16];
  int bi[16];
  #pragma unroll
  for (int i = 0; i < 16; ++i) { bd[i] = FLT_BIG; bi[i] = 0; }

  const int mstart = wv * 1024;
  const unsigned short* rowbase =
      fus + ((size_t)b * MM + mstart + lq) * CC + hi * 8;

  // prime the pipeline with tile 0, chunk 0
  v8bf c0 = *(const v8bf*)(rowbase);
  v8bf c1 = *(const v8bf*)(rowbase + 16);

  for (int tile = 0; tile < 64; ++tile) {
    const int mbase = mstart + tile * 16;
    const unsigned short* row = rowbase + (size_t)tile * 16 * CC;
    if (tile + 2 < 64) __builtin_prefetch(row + 32 * CC, 0, 1);
    v8f acc = {0.f, 0.f, 0.f, 0.f, 0.f, 0.f, 0.f, 0.f};
    #pragma unroll
    for (int kb = 0; kb < CC; kb += 32) {
      // issue loads for the NEXT chunk before consuming the current one
      const unsigned short* np;
      if (kb + 32 < CC)       np = row + kb + 32;
      else if (tile + 1 < 64) np = row + 16 * CC;   // next tile, chunk 0
      else                    np = row;             // clamp: stay in-bounds
      v8bf n0 = *(const v8bf*)np;
      v8bf n1 = *(const v8bf*)(np + 16);
      v16bf at = cmb(c0, c1);
      const unsigned short* qp = &sh_q[lq][kb + hi * 8];
      v16bf bt = ld16(qp, qp + 16);
      acc = wmma_bf(at, bt, acc);
      c0 = n0; c1 = n1;
    }
    #pragma unroll
    for (int r = 0; r < 8; ++r) {
      const int m = mbase + r + hi * 8;
      const float d = r2g[b * MM + m] - 2.0f * acc[r];   // q2 const per query
      if (d < bd[15]) {                                  // sorted-insert
        #pragma unroll
        for (int i = 15; i >= 0; --i) {
          bool pi = d < bd[i];
          bool pm = (i > 0) ? (d < bd[i - 1]) : false;
          float nv = pi ? (pm ? bd[i - 1] : d) : bd[i];
          int   ni = pi ? (pm ? bi[i - 1] : m) : bi[i];
          bd[i] = nv; bi[i] = ni;
        }
      }
    }
  }

  const int mb = lq * 128 + wv * 32 + hi * 16;
  #pragma unroll
  for (int i = 0; i < 16; ++i) { sh_md[mb + i] = bd[i]; sh_mi[mb + i] = bi[i]; }
  __syncthreads();

  if (t < 16) {
    int* op = idx_out + ((size_t)(b * NN) + nbase + t) * KNN;
    for (int j = 0; j < KNN; ++j) {
      float best = FLT_BIG; int bp = 0;
      for (int i = 0; i < 128; ++i) {
        float v = sh_md[t * 128 + i];
        if (v < best) { best = v; bp = i; }
      }
      op[j] = sh_mi[t * 128 + bp];
      sh_md[t * 128 + bp] = FLT_BIG;
    }
  }
}

// --------------------------------------------------------------------------
// Kernel 2: per-point fused pos-MLP + attention MLP + softmax + aggregation.
// One 256-thread (8-wave) block per (b,n). Wave wv owns output channels
// [wv*32, wv*32+32) as two 16-row WMMA tiles; k=16 neighbors are the N dim.
// --------------------------------------------------------------------------
__global__ void k_point(const unsigned short* __restrict__ fus,
                        const int* __restrict__ idx_ws,
                        const float* __restrict__ pcd,
                        const float* __restrict__ pcdb,
                        const float* __restrict__ pos_w1,
                        const float* __restrict__ pos_b1,
                        const float* __restrict__ pos_g,
                        const float* __restrict__ pos_be,
                        const float* __restrict__ pos_m,
                        const float* __restrict__ pos_v,
                        const unsigned short* __restrict__ wpos2,
                        const float* __restrict__ pos_b2,
                        const unsigned short* __restrict__ wat1,
                        const float* __restrict__ attn_b1,
                        const float* __restrict__ attn_g,
                        const float* __restrict__ attn_be,
                        const float* __restrict__ attn_m,
                        const float* __restrict__ attn_v,
                        const unsigned short* __restrict__ wat2,
                        const float* __restrict__ attn_b2,
                        float* __restrict__ out) {
  __shared__ float sh_gf[16][CC];           // group_feat, later gf+pos_emb
  __shared__ float sh_cf[CC];               // center features
  __shared__ float sh_prel[3][16];          // pos_rel
  __shared__ int   sh_idx[16];
  __shared__ unsigned short sh_h[16][PH];   // pos hidden (bf16)
  __shared__ unsigned short sh_X[16][CC];   // qk_rel + pos_emb (bf16)
  __shared__ unsigned short sh_X2[16][CC];  // attn hidden (bf16)
  __shared__ float sh_sA[CC], sh_bA[CC], sh_b2[CC], sh_pb2[CC];
  __shared__ float sh_sP[PH], sh_bP[PH];

  const int n = blockIdx.x, b = blockIdx.y, t = threadIdx.x;
  const int wv = t >> 5, lane = t & 31, lq = lane & 15, hi = lane >> 4;

  // ---- stage A: fold BN params, biases, load idx ----
  {
    float g = attn_g[t], vv = attn_v[t];
    float s = g * rsqrtf(vv + EPSBN);
    sh_sA[t] = s;
    sh_bA[t] = attn_be[t] - attn_m[t] * s + attn_b1[t] * s;
    sh_b2[t] = attn_b2[t];
    sh_pb2[t] = pos_b2[t];
    if (t < PH) {
      float gp = pos_g[t], vp = pos_v[t];
      float sp = gp * rsqrtf(vp + EPSBN);
      sh_sP[t] = sp;
      sh_bP[t] = pos_be[t] - pos_m[t] * sp + pos_b1[t] * sp;
    }
    if (t < 16) sh_idx[t] = idx_ws[((size_t)(b * NN) + n) * KNN + t];
  }
  __syncthreads();

  // ---- stage B: gathers ----
  {
    if (t < 48) {                     // pos_rel = pcd - group_point
      int d = t >> 4, j = t & 15;
      int m = sh_idx[j];
      const float* P = (m < NN) ? pcd : pcdb;
      float gp = P[((size_t)b * 3 + d) * NN + (m & (NN - 1))];
      sh_prel[d][j] = pcd[((size_t)b * 3 + d) * NN + n] - gp;
    }
    int j = t >> 4, cb = (t & 15) * 16;
    const unsigned short* src =
        fus + ((size_t)b * MM + sh_idx[j]) * CC + cb;
    #pragma unroll
    for (int i = 0; i < 16; ++i) sh_gf[j][cb + i] = bf2f(src[i]);
    sh_cf[t] = bf2f(fus[((size_t)b * MM + n) * CC + t]);
  }
  __syncthreads();

  // ---- stage C: pos hidden h = relu(bn(W1*pos_rel + b1)) ----
  {
    int j = t & 15, c0 = (t >> 4) * 4;
    #pragma unroll
    for (int cc = 0; cc < 4; ++cc) {
      int c = c0 + cc;
      float v = pos_w1[c * 3 + 0] * sh_prel[0][j] +
                pos_w1[c * 3 + 1] * sh_prel[1][j] +
                pos_w1[c * 3 + 2] * sh_prel[2][j];
      v = fmaxf(v * sh_sP[c] + sh_bP[c], 0.f);
      sh_h[j][c] = f2bf(v);
    }
  }
  __syncthreads();

  const int rowbase = wv * 32;

  // ---- stage D: pos_embedding GEMM (256x64 @ 64x16), build X and S ----
  {
    v8f acc0 = {0.f,0.f,0.f,0.f,0.f,0.f,0.f,0.f};
    v8f acc1 = acc0;
    #pragma unroll
    for (int kb = 0; kb < PH; kb += 32) {
      const unsigned short* xp = &sh_h[lq][kb + hi * 8];
      v16bf bt = ld16(xp, xp + 16);
      const unsigned short* w0 = wpos2 + (size_t)(rowbase + lq) * PH + kb + hi * 8;
      const unsigned short* w1p = wpos2 + (size_t)(rowbase + 16 + lq) * PH + kb + hi * 8;
      v16bf a0 = ld16(w0, w0 + 16);
      v16bf a1 = ld16(w1p, w1p + 16);
      acc0 = wmma_bf(a0, bt, acc0);
      acc1 = wmma_bf(a1, bt, acc1);
    }
    #pragma unroll
    for (int r = 0; r < 8; ++r) {
      int c = rowbase + r + hi * 8;
      float pe = acc0[r] + sh_pb2[c];
      float g = sh_gf[lq][c];
      sh_X[lq][c] = f2bf(sh_cf[c] - g + pe);
      sh_gf[lq][c] = g + pe;                 // S = group_feat + pos_emb
    }
    #pragma unroll
    for (int r = 0; r < 8; ++r) {
      int c = rowbase + 16 + r + hi * 8;
      float pe = acc1[r] + sh_pb2[c];
      float g = sh_gf[lq][c];
      sh_X[lq][c] = f2bf(sh_cf[c] - g + pe);
      sh_gf[lq][c] = g + pe;
    }
  }
  __syncthreads();

  // ---- stage E: attn1 GEMM (256x256 @ 256x16) + BN + relu ----
  {
    v8f acc0 = {0.f,0.f,0.f,0.f,0.f,0.f,0.f,0.f};
    v8f acc1 = acc0;
    #pragma unroll
    for (int kb = 0; kb < CC; kb += 32) {
      const unsigned short* xp = &sh_X[lq][kb + hi * 8];
      v16bf bt = ld16(xp, xp + 16);
      const unsigned short* w0 = wat1 + (size_t)(rowbase + lq) * CC + kb + hi * 8;
      const unsigned short* w1p = wat1 + (size_t)(rowbase + 16 + lq) * CC + kb + hi * 8;
      v16bf a0 = ld16(w0, w0 + 16);
      v16bf a1 = ld16(w1p, w1p + 16);
      acc0 = wmma_bf(a0, bt, acc0);
      acc1 = wmma_bf(a1, bt, acc1);
    }
    #pragma unroll
    for (int r = 0; r < 8; ++r) {
      int c = rowbase + r + hi * 8;
      float a = fmaxf(acc0[r] * sh_sA[c] + sh_bA[c], 0.f);
      sh_X2[lq][c] = f2bf(a);
    }
    #pragma unroll
    for (int r = 0; r < 8; ++r) {
      int c = rowbase + 16 + r + hi * 8;
      float a = fmaxf(acc1[r] * sh_sA[c] + sh_bA[c], 0.f);
      sh_X2[lq][c] = f2bf(a);
    }
  }
  __syncthreads();

  // ---- stage F: attn2 GEMM + softmax over k (16 lanes) + aggregation ----
  {
    v8f acc0 = {0.f,0.f,0.f,0.f,0.f,0.f,0.f,0.f};
    v8f acc1 = acc0;
    #pragma unroll
    for (int kb = 0; kb < CC; kb += 32) {
      const unsigned short* xp = &sh_X2[lq][kb + hi * 8];
      v16bf bt = ld16(xp, xp + 16);
      const unsigned short* w0 = wat2 + (size_t)(rowbase + lq) * CC + kb + hi * 8;
      const unsigned short* w1p = wat2 + (size_t)(rowbase + 16 + lq) * CC + kb + hi * 8;
      v16bf a0 = ld16(w0, w0 + 16);
      v16bf a1 = ld16(w1p, w1p + 16);
      acc0 = wmma_bf(a0, bt, acc0);
      acc1 = wmma_bf(a1, bt, acc1);
    }
    #pragma unroll
    for (int tt = 0; tt < 2; ++tt) {
      #pragma unroll
      for (int r = 0; r < 8; ++r) {
        int c = rowbase + tt * 16 + r + hi * 8;
        float a = (tt ? acc1[r] : acc0[r]) + sh_b2[c];
        // softmax across the 16 neighbors (lanes within half-wave)
        float mx = a;
        #pragma unroll
        for (int mk = 1; mk < 16; mk <<= 1) mx = fmaxf(mx, __shfl_xor(mx, mk, 32));
        float e = __expf(a - mx);
        float sm = e;
        #pragma unroll
        for (int mk = 1; mk < 16; mk <<= 1) sm += __shfl_xor(sm, mk, 32);
        float w = e / sm;
        float p = w * sh_gf[lq][c];          // w * (group_feat + pos_emb)
        #pragma unroll
        for (int mk = 1; mk < 16; mk <<= 1) p += __shfl_xor(p, mk, 32);
        if (lq == 0) out[((size_t)b * CC + c) * NN + n] = p;
      }
    }
  }
}

// --------------------------------------------------------------------------
extern "C" void kernel_launch(void* const* d_in, const int* in_sizes, int n_in,
                              void* d_out, int out_size, void* d_ws, size_t ws_size,
                              hipStream_t stream) {
  const float* pcd     = (const float*)d_in[0];
  const float* feat    = (const float*)d_in[1];
  const float* pcdb    = (const float*)d_in[2];
  const float* featdb  = (const float*)d_in[3];
  const float* pos_w1  = (const float*)d_in[4];
  const float* pos_b1  = (const float*)d_in[5];
  const float* pos_g   = (const float*)d_in[6];
  const float* pos_be  = (const float*)d_in[7];
  const float* pos_m   = (const float*)d_in[8];
  const float* pos_v   = (const float*)d_in[9];
  const float* pos_w2  = (const float*)d_in[10];
  const float* pos_b2  = (const float*)d_in[11];
  const float* attn_w1 = (const float*)d_in[12];
  const float* attn_b1 = (const float*)d_in[13];
  const float* attn_g  = (const float*)d_in[14];
  const float* attn_be = (const float*)d_in[15];
  const float* attn_m  = (const float*)d_in[16];
  const float* attn_v  = (const float*)d_in[17];
  const float* attn_w2 = (const float*)d_in[18];
  const float* attn_b2 = (const float*)d_in[19];
  float* out = (float*)d_out;

  char* ws = (char*)d_ws;
  unsigned short* fus   = (unsigned short*)(ws);                     // 8 MB
  float*          r2g   = (float*)(ws + 8388608);                    // 64 KB
  int*            idxws = (int*)(ws + 8454144);                      // 512 KB
  unsigned short* wpos2 = (unsigned short*)(ws + 8978432);           // 32 KB
  unsigned short* wat1  = (unsigned short*)(ws + 9011200);           // 128 KB
  unsigned short* wat2  = (unsigned short*)(ws + 9142272);           // 128 KB

  k_prep_feat<<<dim3(MM, BB), 256, 0, stream>>>(feat, featdb, fus, r2g);
  k_cvt_bf16<<<(CC * PH + 255) / 256, 256, 0, stream>>>(pos_w2, wpos2, CC * PH);
  k_cvt_bf16<<<(CC * CC + 255) / 256, 256, 0, stream>>>(attn_w1, wat1, CC * CC);
  k_cvt_bf16<<<(CC * CC + 255) / 256, 256, 0, stream>>>(attn_w2, wat2, CC * CC);
  k_knn<<<dim3(NN / 16, BB), 128, 0, stream>>>(fus, r2g, idxws);
  k_point<<<dim3(NN, BB), 256, 0, stream>>>(
      fus, idxws, pcd, pcdb,
      pos_w1, pos_b1, pos_g, pos_be, pos_m, pos_v, wpos2, pos_b2,
      wat1, attn_b1, attn_g, attn_be, attn_m, attn_v, wat2, attn_b2, out);
}
